// GRUAutoencoder_18339510354796
// MI455X (gfx1250) — compile-verified, matrix operand order checked by reading, not run on hardware
//
#include <hip/hip_runtime.h>
#include <hip/hip_bf16.h>

// ---------------- problem constants ----------------
#define BB 256
#define TT 512
#define CC 64
#define HH 512
#define G3H (3 * HH)   // 1536

typedef unsigned short u16;
typedef __attribute__((ext_vector_type(16))) __bf16 v16bf;
typedef __attribute__((ext_vector_type(8)))  __bf16 v8bf;
typedef __attribute__((ext_vector_type(8)))  float  v8f;

// ---------------- helpers ----------------
__device__ __forceinline__ u16 f2bf(float f) {
    unsigned u = __builtin_bit_cast(unsigned, f);
    u += 0x7FFFu + ((u >> 16) & 1u);              // round-to-nearest-even
    return (u16)(u >> 16);
}

__device__ __forceinline__ float sigf(float x) {
    return 1.0f / (1.0f + __expf(-x));
}

// Load one 32-element bf16 fragment given the per-lane base pointer.
// Per ISA layout the lane needs K=[8*hi,8*hi+8) and K=[16+8*hi,16+8*hi+8),
// which is two contiguous 16B chunks at p and p+16.
__device__ __forceinline__ v16bf ldfrag(const u16* p) {
    union { v16bf v; v8bf h[2]; } u;
    u.h[0] = *(const v8bf*)(p);
    u.h[1] = *(const v8bf*)(p + 16);
    return u.v;
}

__device__ __forceinline__ v8f wmma_bf16(v16bf a, v16bf b, v8f c) {
    return __builtin_amdgcn_wmma_f32_16x16x32_bf16(
        /*neg_a=*/false, a, /*neg_b=*/false, b,
        /*c_mod=*/(short)0, c, /*reuse_a=*/false, /*reuse_b=*/false);
}

// ---------------- f32 -> bf16 weight conversion ----------------
__global__ void cvt_f32_to_bf16(const float* __restrict__ src,
                                u16* __restrict__ dst, int n) {
    for (int i = blockIdx.x * blockDim.x + threadIdx.x; i < n;
         i += gridDim.x * blockDim.x)
        dst[i] = f2bf(src[i]);
}

// ---------------- encoder GRU ----------------
// grid = B/16 workgroups, block = 1024 (32 waves).
// Wave w owns hidden columns [16w, 16w+16). h state kept in registers (WMMA C
// layout, one v8f per wave) + bf16 mirror in LDS (A-matrix source).
#define HLD 520  // padded LDS row stride for h (bf16 elems), keeps 16B align

__global__ void __launch_bounds__(1024)
gru_encoder(const float* __restrict__ x,
            const u16*  __restrict__ wih,   // [1536][64]  bf16
            const u16*  __restrict__ whh,   // [1536][512] bf16
            const float* __restrict__ bih,
            const float* __restrict__ bhh,
            float* __restrict__ henc) {     // [256][512] f32 out
    __shared__ __align__(16) u16 h_bf[16 * HLD];
    __shared__ __align__(16) u16 x_bf[16 * CC];

    const int tid  = threadIdx.x;
    const int w    = tid >> 5;          // 0..31
    const int lane = tid & 31;
    const int b0   = blockIdx.x * 16;
    const int nlo  = lane & 15;
    const int hi8  = (lane >> 4) * 8;
    const int mb   = hi8;               // C-layout row base for this lane
    const int col  = w * 16 + nlo;      // this lane's hidden column

    // per-lane fragment base pointers (bumped by 32 elems per K step)
    const u16* wR0 = whh + (size_t)(col)           * HH + hi8;
    const u16* wZ0 = whh + (size_t)(HH + col)      * HH + hi8;
    const u16* wN0 = whh + (size_t)(2 * HH + col)  * HH + hi8;
    const u16* iR0 = wih + (size_t)(col)           * CC + hi8;
    const u16* iZ0 = wih + (size_t)(HH + col)      * CC + hi8;
    const u16* iN0 = wih + (size_t)(2 * HH + col)  * CC + hi8;
    const u16* ha0 = h_bf + nlo * HLD + hi8;       // A frag (rows = batch)
    const u16* xa0 = x_bf + nlo * CC  + hi8;

    // zero h state
    for (int i = tid; i < 16 * HLD; i += 1024) h_bf[i] = 0;
    v8f hreg = {};

    const float bsR = bih[col]          + bhh[col];
    const float bsZ = bih[HH + col]     + bhh[HH + col];
    const float biN = bih[2 * HH + col];
    const float bhN = bhh[2 * HH + col];
    __syncthreads();

    for (int t = 0; t < TT; ++t) {
        // stage x[b0:b0+16, t, :] as bf16 in LDS
        {
            int i = tid;
            if (i < 16 * CC) {
                int m = i >> 6, c = i & 63;
                x_bf[m * CC + c] =
                    f2bf(x[(size_t)(b0 + m) * TT * CC + t * CC + c]);
            }
        }
        __syncthreads();

        v8f aR = {}, aZ = {}, aNi = {}, aNh = {};

        // input gates: gi = x_t @ Wih.T  (K = 64)
        {
            const u16 *pR = iR0, *pZ = iZ0, *pN = iN0, *pa = xa0;
#pragma clang loop unroll(disable)
            for (int k = 0; k < CC; k += 32) {
                v16bf a = ldfrag(pa);
                aR  = wmma_bf16(a, ldfrag(pR), aR);
                aZ  = wmma_bf16(a, ldfrag(pZ), aZ);
                aNi = wmma_bf16(a, ldfrag(pN), aNi);
                pa += 32; pR += 32; pZ += 32; pN += 32;
                __builtin_amdgcn_sched_barrier(0);
            }
        }
        // hidden gates: gh = h @ Whh.T  (K = 512), Whh streamed from L2
        {
            const u16 *pR = wR0, *pZ = wZ0, *pN = wN0, *pa = ha0;
#pragma clang loop unroll(disable)
            for (int k = 0; k < HH; k += 32) {
                v16bf a = ldfrag(pa);
                __builtin_prefetch(pR + 64, 0, 1);
                aR  = wmma_bf16(a, ldfrag(pR), aR);
                aZ  = wmma_bf16(a, ldfrag(pZ), aZ);
                aNh = wmma_bf16(a, ldfrag(pN), aNh);
                pa += 32; pR += 32; pZ += 32; pN += 32;
                __builtin_amdgcn_sched_barrier(0);
            }
        }
        __syncthreads();   // all reads of h_bf done before rewrite

        for (int i = 0; i < 8; ++i) {
            float r  = sigf(aR[i] + bsR);
            float z  = sigf(aZ[i] + bsZ);
            float hn = aNh[i] + bhN;
            float nn = tanhf(aNi[i] + biN + r * hn);
            float h  = (1.0f - z) * nn + z * hreg[i];
            hreg[i]  = h;
            h_bf[(mb + i) * HLD + col] = f2bf(h);
        }
        __syncthreads();
    }

    // write final hidden state (f32) for the decoder
    for (int i = 0; i < 8; ++i)
        henc[(size_t)(b0 + mb + i) * HH + col] = hreg[i];
}

// ---------------- decoder GRU (free-running) ----------------
#define PLD 64  // LDS row stride for prev token (bf16)

__global__ void __launch_bounds__(1024)
gru_decoder(const float* __restrict__ henc,  // [256][512]
            const u16*  __restrict__ wih,    // [1536][64]  bf16
            const u16*  __restrict__ whh,    // [1536][512] bf16
            const float* __restrict__ bih,
            const float* __restrict__ bhh,
            const u16*  __restrict__ wout,   // [64][512]   bf16
            const float* __restrict__ bout,
            float* __restrict__ out) {       // [256][512][64]
    __shared__ __align__(16) u16 h_bf[16 * HLD];
    __shared__ __align__(16) u16 prev_bf[16 * PLD];

    const int tid  = threadIdx.x;
    const int w    = tid >> 5;
    const int lane = tid & 31;
    const int b0   = blockIdx.x * 16;
    const int nlo  = lane & 15;
    const int hi8  = (lane >> 4) * 8;
    const int mb   = hi8;
    const int col  = w * 16 + nlo;

    const u16* wR0 = whh + (size_t)(col)           * HH + hi8;
    const u16* wZ0 = whh + (size_t)(HH + col)      * HH + hi8;
    const u16* wN0 = whh + (size_t)(2 * HH + col)  * HH + hi8;
    const u16* iR0 = wih + (size_t)(col)           * CC + hi8;
    const u16* iZ0 = wih + (size_t)(HH + col)      * CC + hi8;
    const u16* iN0 = wih + (size_t)(2 * HH + col)  * CC + hi8;
    const u16* ha0 = h_bf + nlo * HLD + hi8;
    const u16* pa0 = prev_bf + nlo * PLD + hi8;

    // start token = zeros
    for (int i = tid; i < 16 * PLD; i += 1024) prev_bf[i] = 0;

    // init h from encoder output, registers + LDS bf16 mirror
    v8f hreg;
    for (int i = 0; i < 8; ++i) {
        float h = henc[(size_t)(b0 + mb + i) * HH + col];
        hreg[i] = h;
        h_bf[(mb + i) * HLD + col] = f2bf(h);
    }
    const float bsR = bih[col]          + bhh[col];
    const float bsZ = bih[HH + col]     + bhh[HH + col];
    const float biN = bih[2 * HH + col];
    const float bhN = bhh[2 * HH + col];
    __syncthreads();

    for (int t = 0; t < TT; ++t) {
        v8f aR = {}, aZ = {}, aNi = {}, aNh = {};

        // gi = prev @ Wih.T  (K = 64, prev from LDS)
        {
            const u16 *pR = iR0, *pZ = iZ0, *pN = iN0, *pa = pa0;
#pragma clang loop unroll(disable)
            for (int k = 0; k < CC; k += 32) {
                v16bf a = ldfrag(pa);
                aR  = wmma_bf16(a, ldfrag(pR), aR);
                aZ  = wmma_bf16(a, ldfrag(pZ), aZ);
                aNi = wmma_bf16(a, ldfrag(pN), aNi);
                pa += 32; pR += 32; pZ += 32; pN += 32;
                __builtin_amdgcn_sched_barrier(0);
            }
        }
        // gh = h @ Whh.T  (K = 512)
        {
            const u16 *pR = wR0, *pZ = wZ0, *pN = wN0, *pa = ha0;
#pragma clang loop unroll(disable)
            for (int k = 0; k < HH; k += 32) {
                v16bf a = ldfrag(pa);
                __builtin_prefetch(pR + 64, 0, 1);
                aR  = wmma_bf16(a, ldfrag(pR), aR);
                aZ  = wmma_bf16(a, ldfrag(pZ), aZ);
                aNh = wmma_bf16(a, ldfrag(pN), aNh);
                pa += 32; pR += 32; pZ += 32; pN += 32;
                __builtin_amdgcn_sched_barrier(0);
            }
        }
        __syncthreads();   // reads of h_bf / prev_bf complete

        for (int i = 0; i < 8; ++i) {
            float r  = sigf(aR[i] + bsR);
            float z  = sigf(aZ[i] + bsZ);
            float hn = aNh[i] + bhN;
            float nn = tanhf(aNi[i] + biN + r * hn);
            float h  = (1.0f - z) * nn + z * hreg[i];
            hreg[i]  = h;
            h_bf[(mb + i) * HLD + col] = f2bf(h);
        }
        __syncthreads();   // h_new visible to projection waves

        // pred = h_new @ Wout.T + bout  (N = 64 -> 4 tiles, waves 0..3)
        if (w < 4) {
            v8f acc = {};
            const u16* pw = wout + (size_t)(w * 16 + nlo) * HH + hi8;
            const u16* pa = ha0;
#pragma clang loop unroll(disable)
            for (int k = 0; k < HH; k += 32) {
                acc = wmma_bf16(ldfrag(pa), ldfrag(pw), acc);
                pa += 32; pw += 32;
                __builtin_amdgcn_sched_barrier(0);
            }
            int oc = w * 16 + nlo;
            float bo = bout[oc];
            for (int i = 0; i < 8; ++i) {
                float p = acc[i] + bo;
                out[(size_t)(b0 + mb + i) * TT * CC + t * CC + oc] = p;
                prev_bf[(mb + i) * PLD + oc] = f2bf(p);
            }
        }
        __syncthreads();   // prev_bf ready for next step
    }
}

// ---------------- host launch ----------------
extern "C" void kernel_launch(void* const* d_in, const int* in_sizes, int n_in,
                              void* d_out, int out_size, void* d_ws, size_t ws_size,
                              hipStream_t stream) {
    const float* x       = (const float*)d_in[0];
    const float* encWih  = (const float*)d_in[1];
    const float* encWhh  = (const float*)d_in[2];
    const float* encBih  = (const float*)d_in[3];
    const float* encBhh  = (const float*)d_in[4];
    const float* decWih  = (const float*)d_in[5];
    const float* decWhh  = (const float*)d_in[6];
    const float* decBih  = (const float*)d_in[7];
    const float* decBhh  = (const float*)d_in[8];
    const float* Wout    = (const float*)d_in[9];
    const float* bout    = (const float*)d_in[10];

    // workspace layout (bf16 weight copies + f32 encoder hidden)
    u16* ws = (u16*)d_ws;
    u16* encWih_bf = ws;                               // 1536*64
    u16* encWhh_bf = encWih_bf + G3H * CC;             // 1536*512
    u16* decWih_bf = encWhh_bf + G3H * HH;
    u16* decWhh_bf = decWih_bf + G3H * CC;
    u16* Wout_bf   = decWhh_bf + G3H * HH;             // 64*512
    float* henc    = (float*)(Wout_bf + CC * HH);      // 256*512 f32

    const int cvtB = 256;
    cvt_f32_to_bf16<<<(G3H * CC + cvtB - 1) / cvtB, cvtB, 0, stream>>>(encWih, encWih_bf, G3H * CC);
    cvt_f32_to_bf16<<<(G3H * HH + cvtB - 1) / cvtB, cvtB, 0, stream>>>(encWhh, encWhh_bf, G3H * HH);
    cvt_f32_to_bf16<<<(G3H * CC + cvtB - 1) / cvtB, cvtB, 0, stream>>>(decWih, decWih_bf, G3H * CC);
    cvt_f32_to_bf16<<<(G3H * HH + cvtB - 1) / cvtB, cvtB, 0, stream>>>(decWhh, decWhh_bf, G3H * HH);
    cvt_f32_to_bf16<<<(CC * HH + cvtB - 1) / cvtB, cvtB, 0, stream>>>(Wout, Wout_bf, CC * HH);

    gru_encoder<<<BB / 16, 1024, 0, stream>>>(x, encWih_bf, encWhh_bf,
                                              encBih, encBhh, henc);
    gru_decoder<<<BB / 16, 1024, 0, stream>>>(henc, decWih_bf, decWhh_bf,
                                              decBih, decBhh, Wout_bf, bout,
                                              (float*)d_out);
}
